// MultiHeadAttention_20160576487681
// MI455X (gfx1250) — compile-verified, matrix operand order checked by reading, not run on hardware
//
#include <hip/hip_runtime.h>

typedef __bf16 bf16_t;
typedef bf16_t v16bf __attribute__((ext_vector_type(16)));
typedef float  v8f   __attribute__((ext_vector_type(8)));
typedef int    v4i   __attribute__((ext_vector_type(4)));
typedef unsigned short us4 __attribute__((ext_vector_type(4)));
typedef unsigned short us8 __attribute__((ext_vector_type(8)));

#define S_LEN  2048
#define D_DIM  1024
#define HEADS  16
#define HD     64

#if __has_builtin(__builtin_amdgcn_global_load_async_to_lds_b128)
#define HAVE_ASYNC 1
#else
#define HAVE_ASYNC 0
#endif

// ---------- helpers ----------

// fp32 -> bf16 bits, round-to-nearest-even
__device__ __forceinline__ unsigned short f2bf(float f) {
    unsigned int u = __builtin_bit_cast(unsigned int, f);
    u = u + 0x7FFFu + ((u >> 16) & 1u);
    return (unsigned short)(u >> 16);
}

__device__ __forceinline__ v8f wmma_bf16(v16bf a, v16bf b, v8f c) {
    return __builtin_amdgcn_wmma_f32_16x16x32_bf16(
        /*neg_a=*/false, a, /*neg_b=*/false, b,
        /*c_mod=*/(short)0, c, /*reuse_a=*/false, /*reuse_b=*/false);
}

// 16-byte global -> LDS copy: async on gfx1250 (ASYNCcnt), VGPR-staged otherwise.
__device__ __forceinline__ void cp16(const unsigned short* g, unsigned short* l) {
#if HAVE_ASYNC
    __builtin_amdgcn_global_load_async_to_lds_b128(
        (__attribute__((address_space(1))) v4i*)(v4i*)g,
        (__attribute__((address_space(3))) v4i*)(v4i*)l,
        0, 0);
#else
    *(us8*)l = *(const us8*)g;
#endif
}

__device__ __forceinline__ void async_drain() {
#if HAVE_ASYNC
#if __has_builtin(__builtin_amdgcn_s_wait_asynccnt)
    __builtin_amdgcn_s_wait_asynccnt(0);
#else
    asm volatile("s_wait_asynccnt 0x0" ::: "memory");
#endif
#endif
}

// Copy a 128x32-half tile g[(row0+i)*D + k0 + j] -> l[i*40 + j] (256 threads).
__device__ __forceinline__ void copy_tile_128x32(
    const unsigned short* __restrict__ g, int row0, int k0,
    unsigned short* l, int tid)
{
    for (int t = tid; t < 512; t += 256) {
        const int i = t >> 2, j = (t & 3) * 8;
        cp16(&g[(size_t)(row0 + i) * D_DIM + k0 + j], &l[i * 40 + j]);
    }
}

// Load a 16x32 fragment from an LDS tile stored [row][K] row-major (ld ushorts).
// Works for A (rows = M) and for B when the source tile is stored [N][K].
// ISA layout: lane l (g=l>>4, m=l&15); element e -> K = e + 8*((e>>3)+g).
__device__ __forceinline__ v16bf load_frag_rowK(const unsigned short* t, int ld, int lane) {
    const int g = lane >> 4;
    const int m = lane & 15;
    v16bf a;
#pragma unroll
    for (int e = 0; e < 16; ++e) {
        const int k = e + (((e >> 3) + g) << 3);
        a[e] = __builtin_bit_cast(bf16_t, t[m * ld + k]);
    }
    return a;
}

// Load a 32x16 B fragment from an LDS tile stored [K][N] row-major (ld ushorts).
__device__ __forceinline__ v16bf load_frag_colK(const unsigned short* t, int ld, int lane) {
    const int g = lane >> 4;
    const int n = lane & 15;
    v16bf b;
#pragma unroll
    for (int e = 0; e < 16; ++e) {
        const int k = e + (((e >> 3) + g) << 3);
        b[e] = __builtin_bit_cast(bf16_t, t[k * ld + n]);
    }
    return b;
}

// ---------- kernel 0: fp32 -> bf16 bulk conversion ----------
__global__ __launch_bounds__(256) void cvt_bf16_kernel(
    const float* __restrict__ src, unsigned short* __restrict__ dst, int n4)
{
    const int i = blockIdx.x * 256 + threadIdx.x;
    if (i < n4) {
        const float4 f = ((const float4*)src)[i];
        us4 o;
        o[0] = f2bf(f.x); o[1] = f2bf(f.y); o[2] = f2bf(f.z); o[3] = f2bf(f.w);
        *(us4*)&dst[(size_t)i * 4] = o;
    }
}

// ---------- kernel 1: fused QKV projection (y = x @ W^T [+ b]) ----------
// 128x128 block tile, Kstep=32, double-buffered async LDS tiles.
// grid: (M/128, N/128, 3), block 256. Output bf16 in [B,H,S,hd] layout.
__global__ __launch_bounds__(256) void qkv_proj_kernel(
    const unsigned short* __restrict__ Xb,
    const unsigned short* __restrict__ Wqb, const unsigned short* __restrict__ Wkb,
    const unsigned short* __restrict__ Wvb, const float* __restrict__ bv,
    unsigned short* __restrict__ Qh, unsigned short* __restrict__ Kh,
    unsigned short* __restrict__ Vh)
{
    __shared__ unsigned short As[2][128 * 40];
    __shared__ unsigned short Bs[2][128 * 40];

    const int tid  = threadIdx.x;
    const int wave = tid >> 5;
    const int lane = tid & 31;
    const int mi0  = (wave & 3) * 2;   // 2 of 8 M-subtiles
    const int ni0  = (wave >> 2) * 4;  // 4 of 8 N-subtiles

    const int row0  = blockIdx.x * 128;  // over M = B*S
    const int n0    = blockIdx.y * 128;  // over N = D
    const int which = blockIdx.z;        // 0=Q 1=K 2=V

    const unsigned short* W = (which == 0) ? Wqb : (which == 1) ? Wkb : Wvb;

    v8f acc[2][4] = {};

    copy_tile_128x32(Xb, row0, 0, As[0], tid);
    copy_tile_128x32(W,  n0,   0, Bs[0], tid);
    async_drain();
    __syncthreads();

    for (int k0 = 0; k0 < D_DIM; k0 += 32) {
        const int cur = (k0 >> 5) & 1;
        if (k0 + 32 < D_DIM) {  // overlap next tile load with WMMA on current
            copy_tile_128x32(Xb, row0, k0 + 32, As[cur ^ 1], tid);
            copy_tile_128x32(W,  n0,   k0 + 32, Bs[cur ^ 1], tid);
        }

        v16bf a[2];
#pragma unroll
        for (int m = 0; m < 2; ++m)
            a[m] = load_frag_rowK(&As[cur][(mi0 + m) * 16 * 40], 40, lane);
#pragma unroll
        for (int nn = 0; nn < 4; ++nn) {
            const v16bf b = load_frag_rowK(&Bs[cur][(ni0 + nn) * 16 * 40], 40, lane);
#pragma unroll
            for (int m = 0; m < 2; ++m)
                acc[m][nn] = wmma_bf16(a[m], b, acc[m][nn]);
        }

        async_drain();
        __syncthreads();
    }

    const int g = lane >> 4, n = lane & 15;
    unsigned short* outp = (which == 0) ? Qh : (which == 1) ? Kh : Vh;
#pragma unroll
    for (int m = 0; m < 2; ++m) {
#pragma unroll
        for (int nn = 0; nn < 4; ++nn) {
            const int col = n0 + (ni0 + nn) * 16 + n;   // global d
            const int h = col >> 6, dh = col & 63;
            const float bias = (which == 2) ? bv[col] : 0.0f;
#pragma unroll
            for (int r = 0; r < 8; ++r) {
                const int R = row0 + (mi0 + m) * 16 + r + 8 * g;  // row in B*S
                const int bb = R >> 11, s = R & (S_LEN - 1);
                outp[((size_t)(bb * HEADS + h) * S_LEN + s) * HD + dh] =
                    f2bf(acc[m][nn][r] + bias);
            }
        }
    }
}

// ---------- kernel 2: flash attention with causal mask ----------
// grid: (S/64, H, B), block 256. Q/K/V bf16 [B,H,S,hd] -> ctx bf16 [B,S,D].
__global__ __launch_bounds__(256) void attn_kernel(
    const unsigned short* __restrict__ Qb,
    const unsigned short* __restrict__ Kb,
    const unsigned short* __restrict__ Vb,
    unsigned short* __restrict__ Cb)
{
    __shared__ unsigned short Qs[64 * 72];
    __shared__ unsigned short Ks[64 * 72];
    __shared__ unsigned short Vs[64 * 72];
    __shared__ unsigned short Pb[64 * 72];
    __shared__ float Ssc[64 * 68];
    __shared__ float mrow[64], lrow[64], crow[64];

    const int tid  = threadIdx.x;
    const int wave = tid >> 5;
    const int lane = tid & 31;
    const int mi   = wave >> 1;
    const int ni0  = (wave & 1) * 2;
    const int g    = lane >> 4;
    const int n    = lane & 15;

    const int qt = blockIdx.x;
    const int h  = blockIdx.y;
    const int bb = blockIdx.z;
    const int q0 = qt * 64;
    const size_t headoff = (size_t)(bb * HEADS + h) * S_LEN * HD;

    // load Q tile (64x64 halves = 512 x 16B)
    for (int t = tid; t < 512; t += 256) {
        const int i = t >> 3, d = (t & 7) * 8;
        cp16(&Qb[headoff + (size_t)(q0 + i) * HD + d], &Qs[i * 72 + d]);
    }
    if (tid < 64) { mrow[tid] = -3.0e38f; lrow[tid] = 0.0f; }

    v8f o[2] = {};

    for (int j = 0; j <= qt; ++j) {
        const int k0 = j * 64;
        __syncthreads();
        for (int t = tid; t < 512; t += 256) {
            const int i = t >> 3, d = (t & 7) * 8;
            cp16(&Kb[headoff + (size_t)(k0 + i) * HD + d], &Ks[i * 72 + d]);
            cp16(&Vb[headoff + (size_t)(k0 + i) * HD + d], &Vs[i * 72 + d]);
        }
        async_drain();
        __syncthreads();

        // S = (Q K^T) / sqrt(hd), masked on diagonal block
        {
            v8f sc[2] = {};
            for (int kc = 0; kc < HD; kc += 32) {
                const v16bf a = load_frag_rowK(&Qs[mi * 16 * 72 + kc], 72, lane);
#pragma unroll
                for (int t = 0; t < 2; ++t) {
                    const v16bf bf = load_frag_rowK(&Ks[(ni0 + t) * 16 * 72 + kc], 72, lane);
                    sc[t] = wmma_bf16(a, bf, sc[t]);
                }
            }
#pragma unroll
            for (int t = 0; t < 2; ++t) {
                const int key = k0 + (ni0 + t) * 16 + n;
#pragma unroll
                for (int r = 0; r < 8; ++r) {
                    const int qrow = mi * 16 + r + 8 * g;
                    float sv = sc[t][r] * 0.125f;            // 1/sqrt(64)
                    if (key > q0 + qrow) sv = -3.0e38f;      // causal mask
                    Ssc[qrow * 68 + (ni0 + t) * 16 + n] = sv;
                }
            }
        }
        __syncthreads();

        // online softmax row pass
        if (tid < 64) {
            const int row = tid;
            const float mo = mrow[row];
            float mx = mo;
            for (int c = 0; c < 64; ++c) {
                const float v = Ssc[row * 68 + c];
                mx = v > mx ? v : mx;
            }
            const float corr = __expf(mo - mx);
            float sum = 0.0f;
            for (int c = 0; c < 64; ++c) {
                const float p = __expf(Ssc[row * 68 + c] - mx);
                sum += p;
                Pb[row * 72 + c] = f2bf(p);
            }
            mrow[row] = mx;
            lrow[row] = lrow[row] * corr + sum;
            crow[row] = corr;
        }
        __syncthreads();

        // rescale O and accumulate O += P @ V
#pragma unroll
        for (int t = 0; t < 2; ++t)
#pragma unroll
            for (int r = 0; r < 8; ++r)
                o[t][r] *= crow[mi * 16 + r + 8 * g];

        for (int kc = 0; kc < 64; kc += 32) {
            const v16bf a = load_frag_rowK(&Pb[mi * 16 * 72 + kc], 72, lane);
#pragma unroll
            for (int t = 0; t < 2; ++t) {
                const int ni = ni0 + t;
                const v16bf bf = load_frag_colK(&Vs[kc * 72 + ni * 16], 72, lane);
                o[t] = wmma_bf16(a, bf, o[t]);
            }
        }
    }
    __syncthreads();

    // normalize and write ctx (bf16, [B,S,D] with D = h*64 + d)
#pragma unroll
    for (int t = 0; t < 2; ++t) {
        const int ni = ni0 + t;
#pragma unroll
        for (int r = 0; r < 8; ++r) {
            const int qrow = mi * 16 + r + 8 * g;
            const float val = o[t][r] / lrow[qrow];
            const size_t idx =
                (size_t)(bb * S_LEN + q0 + qrow) * D_DIM + h * HD + ni * 16 + n;
            Cb[idx] = f2bf(val);
        }
    }
}

// ---------- kernel 3: output projection out = ctx @ Wo^T + bo (fp32 out) ----------
__global__ __launch_bounds__(256) void oproj_kernel(
    const unsigned short* __restrict__ Cbuf,
    const unsigned short* __restrict__ Wob, const float* __restrict__ bo,
    float* __restrict__ out)
{
    __shared__ unsigned short As[2][128 * 40];
    __shared__ unsigned short Bs[2][128 * 40];

    const int tid  = threadIdx.x;
    const int wave = tid >> 5;
    const int lane = tid & 31;
    const int mi0  = (wave & 3) * 2;
    const int ni0  = (wave >> 2) * 4;

    const int row0 = blockIdx.x * 128;
    const int n0   = blockIdx.y * 128;

    v8f acc[2][4] = {};

    copy_tile_128x32(Cbuf, row0, 0, As[0], tid);
    copy_tile_128x32(Wob,  n0,   0, Bs[0], tid);
    async_drain();
    __syncthreads();

    for (int k0 = 0; k0 < D_DIM; k0 += 32) {
        const int cur = (k0 >> 5) & 1;
        if (k0 + 32 < D_DIM) {
            copy_tile_128x32(Cbuf, row0, k0 + 32, As[cur ^ 1], tid);
            copy_tile_128x32(Wob,  n0,   k0 + 32, Bs[cur ^ 1], tid);
        }

        v16bf a[2];
#pragma unroll
        for (int m = 0; m < 2; ++m)
            a[m] = load_frag_rowK(&As[cur][(mi0 + m) * 16 * 40], 40, lane);
#pragma unroll
        for (int nn = 0; nn < 4; ++nn) {
            const v16bf b = load_frag_rowK(&Bs[cur][(ni0 + nn) * 16 * 40], 40, lane);
#pragma unroll
            for (int m = 0; m < 2; ++m)
                acc[m][nn] = wmma_bf16(a[m], b, acc[m][nn]);
        }

        async_drain();
        __syncthreads();
    }

    const int g = lane >> 4, n = lane & 15;
#pragma unroll
    for (int m = 0; m < 2; ++m) {
#pragma unroll
        for (int nn = 0; nn < 4; ++nn) {
            const int col = n0 + (ni0 + nn) * 16 + n;
            const float bias = bo[col];
#pragma unroll
            for (int r = 0; r < 8; ++r) {
                const int R = row0 + (mi0 + m) * 16 + r + 8 * g;
                out[(size_t)R * D_DIM + col] = acc[m][nn][r] + bias;
            }
        }
    }
}

// ---------- host entry ----------
extern "C" void kernel_launch(void* const* d_in, const int* in_sizes, int n_in,
                              void* d_out, int out_size, void* d_ws, size_t ws_size,
                              hipStream_t stream) {
    (void)in_sizes; (void)n_in; (void)out_size; (void)ws_size;

    const float* x  = (const float*)d_in[0];
    const float* Wq = (const float*)d_in[1];
    const float* Wk = (const float*)d_in[2];
    const float* Wv = (const float*)d_in[3];
    const float* bv = (const float*)d_in[4];
    const float* Wo = (const float*)d_in[5];
    const float* bo = (const float*)d_in[6];
    float* out = (float*)d_out;

    // workspace layout (bf16 elements):
    const size_t XN = (size_t)2 * S_LEN * D_DIM;   // 4,194,304 (x, per-head bufs, ctx)
    const size_t WN = (size_t)D_DIM * D_DIM;       // 1,048,576
    unsigned short* Xb  = (unsigned short*)d_ws;   // [B*S, D]
    unsigned short* Wqb = Xb  + XN;
    unsigned short* Wkb = Wqb + WN;
    unsigned short* Wvb = Wkb + WN;
    unsigned short* Wob = Wvb + WN;
    unsigned short* Qh  = Wob + WN;                // [B,H,S,hd]
    unsigned short* Kh  = Qh  + XN;
    unsigned short* Vh  = Kh  + XN;
    unsigned short* Cb  = Vh  + XN;                // [B,S,D]

    // 0) bulk fp32->bf16 conversions (one pass each)
    cvt_bf16_kernel<<<(int)(XN / 4 / 256), 256, 0, stream>>>(x,  Xb,  (int)(XN / 4));
    cvt_bf16_kernel<<<(int)(WN / 4 / 256), 256, 0, stream>>>(Wq, Wqb, (int)(WN / 4));
    cvt_bf16_kernel<<<(int)(WN / 4 / 256), 256, 0, stream>>>(Wk, Wkb, (int)(WN / 4));
    cvt_bf16_kernel<<<(int)(WN / 4 / 256), 256, 0, stream>>>(Wv, Wvb, (int)(WN / 4));
    cvt_bf16_kernel<<<(int)(WN / 4 / 256), 256, 0, stream>>>(Wo, Wob, (int)(WN / 4));

    // 1) QKV projections: M = B*S = 4096, N = D = 1024
    qkv_proj_kernel<<<dim3(4096 / 128, D_DIM / 128, 3), 256, 0, stream>>>(
        Xb, Wqb, Wkb, Wvb, bv, Qh, Kh, Vh);

    // 2) flash attention per (q-tile, head, batch)
    attn_kernel<<<dim3(S_LEN / 64, HEADS, 2), 256, 0, stream>>>(Qh, Kh, Vh, Cb);

    // 3) output projection -> fp32 d_out
    oproj_kernel<<<dim3(4096 / 128, D_DIM / 128, 1), 256, 0, stream>>>(Cb, Wob, bo, out);
}
